// TensorParallelQwenAttention_20495583936681
// MI455X (gfx1250) — matrix operand order repaired and checked by reading, not compiled, not run on hardware
//
#include <hip/hip_runtime.h>

// Problem constants (match reference)
#define B_   2
#define S_   2048
#define HID_ 2048
#define NH_  16
#define NKV_ 4
#define HD_  128
#define M_   (B_ * S_)   // 4096 rows

typedef __attribute__((ext_vector_type(16))) __bf16       v16bf;
typedef __attribute__((ext_vector_type(8)))  float        v8f;
typedef __attribute__((ext_vector_type(4)))  float        f32x4;
typedef __attribute__((ext_vector_type(4)))  unsigned int u32x4;

union BF16Frag { v16bf v; u32x4 q[2]; };

static __device__ __forceinline__ v8f v8f_zero() {
    v8f z;
#pragma unroll
    for (int i = 0; i < 8; ++i) z[i] = 0.0f;
    return z;
}

static __device__ __forceinline__ v8f wmma_bf16(const BF16Frag& a, const BF16Frag& b, v8f c) {
    // D = A(16x32 bf16) * B(32x16 bf16) + C(16x16 f32)
    return __builtin_amdgcn_wmma_f32_16x16x32_bf16(false, a.v, false, b.v, (short)0, c, false, false);
}

// ---------------------------------------------------------------------------
// Generic WMMA GEMM: C[M,N] = A[M,K] @ W[K,N] (+ bias[N])
//   A_BF16:   A is bf16 (else fp32, converted while staging)
//   OUT_BF16: C is bf16 (else fp32)
// Workgroup: 256 threads = 8 waves; tile 128(M) x 128(N), BK = 32.
// Wave wid: wm = wid&3 -> 32 rows, wn = wid>>2 -> 64 cols (2x4 sub-tiles of 16x16).
// ---------------------------------------------------------------------------
template <bool A_BF16, bool OUT_BF16, bool HAS_BIAS>
__global__ __launch_bounds__(256) void gemm_bf16_wmma(const void*  __restrict__ Ap,
                                                      const float* __restrict__ W,
                                                      const float* __restrict__ bias,
                                                      void*        __restrict__ Cp,
                                                      int M, int N, int K) {
    __shared__ __bf16 lA[128][40];   // row-major [m][k], pitch 40 (80B, 16B-aligned)
    __shared__ __bf16 lB[128][40];   // transposed [n][k]

    const int n0   = blockIdx.x * 128;
    const int m0   = blockIdx.y * 128;
    const int t    = threadIdx.x;
    const int lane = t & 31, ln = lane & 15, half = lane >> 4;
    const int wid  = t >> 5;
    const int wm   = wid & 3;   // 0..3
    const int wn   = wid >> 2;  // 0..1

    v8f acc[2][4];
#pragma unroll
    for (int i = 0; i < 2; ++i)
#pragma unroll
        for (int j = 0; j < 4; ++j) acc[i][j] = v8f_zero();

    // Staging assignments: A tile 128x32 (thread -> 16 elems), B tile 32x128.
    const int arow = t >> 1;
    const int akq  = (t & 1) << 4;   // 0 or 16
    const int bk   = t >> 3;         // 0..31
    const int bnq  = (t & 7) << 4;   // 0..112

    for (int k0 = 0; k0 < K; k0 += 32) {
        __syncthreads();
        // ---- stage A (convert fp32 -> bf16 if needed) ----
        if (A_BF16) {
            const __bf16* ap = (const __bf16*)Ap + (size_t)(m0 + arow) * K + k0 + akq;
            *(u32x4*)&lA[arow][akq + 0] = ((const u32x4*)ap)[0];
            *(u32x4*)&lA[arow][akq + 8] = ((const u32x4*)ap)[1];
        } else {
            const float* ap = (const float*)Ap + (size_t)(m0 + arow) * K + k0 + akq;
            __bf16 tmp[16] __attribute__((aligned(16)));
#pragma unroll
            for (int j = 0; j < 16; j += 4) {
                f32x4 f = *(const f32x4*)(ap + j);
                tmp[j + 0] = (__bf16)f.x; tmp[j + 1] = (__bf16)f.y;
                tmp[j + 2] = (__bf16)f.z; tmp[j + 3] = (__bf16)f.w;
            }
            *(u32x4*)&lA[arow][akq + 0] = ((const u32x4*)tmp)[0];
            *(u32x4*)&lA[arow][akq + 8] = ((const u32x4*)tmp)[1];
        }
        // ---- stage W transposed: lB[n][k] = W[k0+k][n0+n] ----
        {
            const float* wp = W + (size_t)(k0 + bk) * N + n0 + bnq;
#pragma unroll
            for (int j = 0; j < 16; j += 4) {
                f32x4 f = *(const f32x4*)(wp + j);
                lB[bnq + j + 0][bk] = (__bf16)f.x;
                lB[bnq + j + 1][bk] = (__bf16)f.y;
                lB[bnq + j + 2][bk] = (__bf16)f.z;
                lB[bnq + j + 3][bk] = (__bf16)f.w;
            }
            if (k0 + 32 < K) __builtin_prefetch(wp + (size_t)32 * N, 0, 1);
        }
        __syncthreads();

        // ---- fragments (contiguous k = 16*half + s, same in A and B) ----
        BF16Frag af[2], bf[4];
#pragma unroll
        for (int sm = 0; sm < 2; ++sm) {
            const __bf16* p = &lA[(wm << 5) + (sm << 4) + ln][half << 4];
            af[sm].q[0] = *(const u32x4*)p;
            af[sm].q[1] = *(const u32x4*)(p + 8);
        }
#pragma unroll
        for (int sn = 0; sn < 4; ++sn) {
            const __bf16* p = &lB[(wn << 6) + (sn << 4) + ln][half << 4];
            bf[sn].q[0] = *(const u32x4*)p;
            bf[sn].q[1] = *(const u32x4*)(p + 8);
        }
#pragma unroll
        for (int sm = 0; sm < 2; ++sm)
#pragma unroll
            for (int sn = 0; sn < 4; ++sn)
                acc[sm][sn] = wmma_bf16(af[sm], bf[sn], acc[sm][sn]);
    }

    // ---- epilogue: C VGPR r -> row (r + 8*half), lane ln -> col ----
#pragma unroll
    for (int sm = 0; sm < 2; ++sm) {
#pragma unroll
        for (int sn = 0; sn < 4; ++sn) {
            const int n = n0 + (wn << 6) + (sn << 4) + ln;
            float bv = 0.0f;
            if (HAS_BIAS) bv = bias[n];
#pragma unroll
            for (int r = 0; r < 8; ++r) {
                const int m = m0 + (wm << 5) + (sm << 4) + r + (half << 3);
                const float v = acc[sm][sn][r] + bv;
                if (OUT_BF16) ((__bf16*)Cp)[(size_t)m * N + n] = (__bf16)v;
                else          ((float*)Cp)[(size_t)m * N + n]  = v;
            }
        }
    }
}

// ---------------------------------------------------------------------------
// RoPE in place on bf16 [rows][heads*128]; pair (2*d2, 2*d2+1) rotated.
// ---------------------------------------------------------------------------
__global__ void rope_kernel(__bf16* __restrict__ X, int heads, int total) {
    const int idx = blockIdx.x * blockDim.x + threadIdx.x;
    if (idx >= total) return;
    const int d2   = idx & 63;
    const int tmp  = idx >> 6;
    const int head = tmp % heads;
    const int row  = tmp / heads;
    const int s    = row & (S_ - 1);   // position in sequence
    // freq = theta^(-2*d2/HD), theta = 10000 -> ln = 9.2103403719...
    const float freq = __expf(-9.210340371976184f * (float)(2 * d2) * (1.0f / 128.0f));
    const float ang  = (float)s * freq;
    float sn, c;
    __sincosf(ang, &sn, &c);
    const size_t base = ((size_t)row * heads + head) * HD_ + 2 * d2;
    const float x1 = (float)X[base], x2 = (float)X[base + 1];
    X[base]     = (__bf16)(x1 * c - x2 * sn);
    X[base + 1] = (__bf16)(x1 * sn + x2 * c);
}

// ---------------------------------------------------------------------------
// V transpose: V[b*S+s][kvh*128+d] -> Vt[(b*NKV+kvh)*128+d][s]
// ---------------------------------------------------------------------------
__global__ void vtrans_kernel(const __bf16* __restrict__ V, __bf16* __restrict__ Vt) {
    const int idx = blockIdx.x * blockDim.x + threadIdx.x;   // < M_*512
    const int c   = idx & 511;
    const int row = idx >> 9;
    const int b   = row >> 11;
    const int s   = row & 2047;
    Vt[((size_t)(b * (NKV_ * HD_) + c)) * S_ + s] = V[idx];
}

// ---------------------------------------------------------------------------
// Flash attention (causal, GQA rep=4). One wave per (b, h, 16-row q tile).
// QK^T: A = Q (d-contig), B = K rows (d-contig) -> direct B-fragment loads.
// PV:   A = P via LDS C->A relayout, B = Vt rows (key-contig).
// ---------------------------------------------------------------------------
__global__ __launch_bounds__(32) void flash_attn_kernel(const __bf16* __restrict__ Q,
                                                        const __bf16* __restrict__ Kc,
                                                        const __bf16* __restrict__ Vt,
                                                        __bf16*       __restrict__ ctx) {
    __shared__ __bf16 lP[16][40];

    const int qb  = blockIdx.x;      // 0..S/16-1
    const int h   = blockIdx.y;      // 0..15
    const int b   = blockIdx.z;      // 0..1
    const int kvh = h >> 2;          // GQA: 4 Q heads per KV head
    const int lane = threadIdx.x & 31, ln = lane & 15, half = lane >> 4;
    const float scale = 0.08838834764831845f;   // 1/sqrt(128)

    // Resident Q fragments: 16 rows x 128 d = 4 k-steps of 32
    BF16Frag qf[4];
    {
        const __bf16* qp = Q + ((size_t)(b * S_ + qb * 16 + ln)) * (NH_ * HD_) + h * HD_;
#pragma unroll
        for (int ds = 0; ds < 4; ++ds) {
            const __bf16* p = qp + ds * 32 + (half << 4);
            qf[ds].q[0] = *(const u32x4*)p;
            qf[ds].q[1] = *(const u32x4*)(p + 8);
        }
    }

    v8f oacc[8];
#pragma unroll
    for (int i = 0; i < 8; ++i) oacc[i] = v8f_zero();
    float mrun[8], lrun[8];
#pragma unroll
    for (int r = 0; r < 8; ++r) { mrun[r] = -1e30f; lrun[r] = 0.0f; }

    const int qend = qb * 16 + 15;
    const int nblk = qend / 32 + 1;   // 32-key blocks (causal)

    for (int kb = 0; kb < nblk; ++kb) {
        const int key0 = kb * 32;
        v8f s0 = v8f_zero(), s1 = v8f_zero();
        {
            BF16Frag kf;
            const __bf16* kp0 = Kc + ((size_t)(b * S_ + key0 + ln))      * (NKV_ * HD_) + kvh * HD_;
            const __bf16* kp1 = Kc + ((size_t)(b * S_ + key0 + 16 + ln)) * (NKV_ * HD_) + kvh * HD_;
#pragma unroll
            for (int ds = 0; ds < 4; ++ds) {
                const __bf16* p = kp0 + ds * 32 + (half << 4);
                kf.q[0] = *(const u32x4*)p; kf.q[1] = *(const u32x4*)(p + 8);
                s0 = wmma_bf16(qf[ds], kf, s0);
                p = kp1 + ds * 32 + (half << 4);
                kf.q[0] = *(const u32x4*)p; kf.q[1] = *(const u32x4*)(p + 8);
                s1 = wmma_bf16(qf[ds], kf, s1);
            }
        }
        // ---- online softmax (row = r + 8*half, col = ln within 16-lane group) ----
        float alpha[8];
#pragma unroll
        for (int r = 0; r < 8; ++r) {
            const int qrow = qb * 16 + r + (half << 3);
            float a  = s0[r] * scale; if (key0 + ln > qrow)      a  = -3e30f;
            float bb = s1[r] * scale; if (key0 + 16 + ln > qrow) bb = -3e30f;
            float mx = fmaxf(a, bb);
#pragma unroll
            for (int off = 8; off >= 1; off >>= 1) mx = fmaxf(mx, __shfl_xor(mx, off, 32));
            const float mnew = fmaxf(mrun[r], mx);
            alpha[r] = __expf(mrun[r] - mnew);
            const float p0 = __expf(a - mnew);
            const float p1 = __expf(bb - mnew);
            float rs = p0 + p1;
#pragma unroll
            for (int off = 8; off >= 1; off >>= 1) rs += __shfl_xor(rs, off, 32);
            lrun[r] = lrun[r] * alpha[r] + rs;
            mrun[r] = mnew;
            lP[r + (half << 3)][ln]      = (__bf16)p0;   // C-layout -> LDS 16x32 tile
            lP[r + (half << 3)][16 + ln] = (__bf16)p1;
        }
#pragma unroll
        for (int dt = 0; dt < 8; ++dt)
#pragma unroll
            for (int r = 0; r < 8; ++r) oacc[dt][r] *= alpha[r];

        // ---- P as A-fragment (16x32) from LDS; same-wave DS ops are in-order ----
        BF16Frag pf;
        {
            const __bf16* p = &lP[ln][half << 4];
            pf.q[0] = *(const u32x4*)p;
            pf.q[1] = *(const u32x4*)(p + 8);
        }
        // ---- PV: 8 d-subtiles, B from Vt (key-contiguous rows) ----
#pragma unroll
        for (int dt = 0; dt < 8; ++dt) {
            BF16Frag vf;
            const __bf16* vp = Vt + ((size_t)((b * NKV_ + kvh) * HD_ + dt * 16 + ln)) * S_
                                  + key0 + (half << 4);
            vf.q[0] = *(const u32x4*)vp;
            vf.q[1] = *(const u32x4*)(vp + 8);
            oacc[dt] = wmma_bf16(pf, vf, oacc[dt]);
        }
    }

    // ---- finalize: O /= l, write ctx[b, q, h, d] in bf16 ----
#pragma unroll
    for (int r = 0; r < 8; ++r) {
        const float inv = 1.0f / lrun[r];
        const int qrow  = qb * 16 + r + (half << 3);
        __bf16* op = ctx + ((size_t)(b * S_ + qrow)) * (NH_ * HD_) + h * HD_;
#pragma unroll
        for (int dt = 0; dt < 8; ++dt)
            op[dt * 16 + ln] = (__bf16)(oacc[dt][r] * inv);
    }
}

// ---------------------------------------------------------------------------
extern "C" void kernel_launch(void* const* d_in, const int* in_sizes, int n_in,
                              void* d_out, int out_size, void* d_ws, size_t ws_size,
                              hipStream_t stream) {
    (void)in_sizes; (void)n_in; (void)out_size; (void)ws_size;
    const float* hid = (const float*)d_in[0];
    const float* q_w = (const float*)d_in[1];
    const float* q_b = (const float*)d_in[2];
    const float* k_w = (const float*)d_in[3];
    const float* k_b = (const float*)d_in[4];
    const float* v_w = (const float*)d_in[5];
    const float* v_b = (const float*)d_in[6];
    const float* o_w = (const float*)d_in[7];
    float* out = (float*)d_out;

    char* ws = (char*)d_ws;
    size_t off = 0;
    auto carve = [&](size_t bytes) -> void* {
        void* p = ws + off;
        off = (off + bytes + 255) & ~(size_t)255;
        return p;
    };
    __bf16* qbuf = (__bf16*)carve((size_t)M_ * NH_  * HD_ * 2);  // 16 MB
    __bf16* kbuf = (__bf16*)carve((size_t)M_ * NKV_ * HD_ * 2);  //  4 MB
    __bf16* vbuf = (__bf16*)carve((size_t)M_ * NKV_ * HD_ * 2);  //  4 MB
    __bf16* vtb  = (__bf16*)carve((size_t)M_ * NKV_ * HD_ * 2);  //  4 MB
    __bf16* ctx  = (__bf16*)carve((size_t)M_ * NH_  * HD_ * 2);  // 16 MB

    const dim3 blk(256);
    // QKV projections (fp32 A -> bf16 out, + bias)
    gemm_bf16_wmma<false, true, true><<<dim3((NH_ * HD_) / 128, M_ / 128), blk, 0, stream>>>(
        hid, q_w, q_b, qbuf, M_, NH_ * HD_, HID_);
    gemm_bf16_wmma<false, true, true><<<dim3((NKV_ * HD_) / 128, M_ / 128), blk, 0, stream>>>(
        hid, k_w, k_b, kbuf, M_, NKV_ * HD_, HID_);
    gemm_bf16_wmma<false, true, true><<<dim3((NKV_ * HD_) / 128, M_ / 128), blk, 0, stream>>>(
        hid, v_w, v_b, vbuf, M_, NKV_ * HD_, HID_);
    // RoPE
    {
        const int totq = M_ * NH_ * 64;
        rope_kernel<<<(totq + 255) / 256, blk, 0, stream>>>(qbuf, NH_, totq);
        const int totk = M_ * NKV_ * 64;
        rope_kernel<<<(totk + 255) / 256, blk, 0, stream>>>(kbuf, NKV_, totk);
    }
    // V transpose for PV B-fragments
    {
        const int tot = M_ * NKV_ * HD_;
        vtrans_kernel<<<(tot + 255) / 256, blk, 0, stream>>>(vbuf, vtb);
    }
    // Flash attention
    flash_attn_kernel<<<dim3(S_ / 16, NH_, B_), dim3(32), 0, stream>>>(qbuf, kbuf, vtb, ctx);
    // Output projection (bf16 A -> fp32 out, no bias)
    gemm_bf16_wmma<true, false, false><<<dim3(HID_ / 128, M_ / 128), blk, 0, stream>>>(
        ctx, o_w, nullptr, out, M_, HID_, NH_ * HD_);
}
